// DualCodebookQuantizer_32315333935399
// MI455X (gfx1250) — compile-verified, matrix operand order checked by reading, not run on hardware
//
#include <hip/hip_runtime.h>
#include <hip/hip_bf16.h>

typedef _Float16 v8h  __attribute__((ext_vector_type(8)));
typedef _Float16 v16h __attribute__((ext_vector_type(16)));
typedef float    v8f  __attribute__((ext_vector_type(8)));

#define BB 16
#define SS 4096
#define DD 512
#define KK 256

static constexpr long long OUT_Q    = 0;
static constexpr long long OUT_LOSS = (long long)BB * SS * DD;                 // 33554432
static constexpr long long OUT_IDX  = OUT_LOSS + 1;                            // 33554433
static constexpr long long OUT_W    = OUT_IDX + (long long)BB * SS;            // 33619969

// ws layout (floats)
#define WS_POOLED 0
#define WS_C2     8192
#define WS_W      8448
#define WS_CBH    8480   // _Float16[256*512] lives here (16B aligned: 8480*4 = 33920)

#define XH_STRIDE 520    // 512 + 8 halves pad -> conflict-free b128 LDS loads
#define SC_STRIDE 260    // 256 + 4 floats pad

// ---------------------------------------------------------------- init
__global__ void zero_kernel(float* __restrict__ pooled, float* __restrict__ lossSlot) {
  int tid = threadIdx.x;
  for (int i = tid; i < BB * DD; i += 256) pooled[i] = 0.0f;
  if (tid == 0) *lossSlot = 0.0f;
}

// ---------------------------------------------------------------- codebook -> f16 + norms
__global__ __launch_bounds__(256) void prep_cb(const float* __restrict__ cb,
                                               _Float16* __restrict__ cbh,
                                               float* __restrict__ c2) {
  __shared__ float red[256];
  int k = blockIdx.x, tid = threadIdx.x;
  float v0 = cb[k * DD + tid];
  float v1 = cb[k * DD + tid + 256];
  cbh[k * DD + tid]       = (_Float16)v0;
  cbh[k * DD + tid + 256] = (_Float16)v1;
  red[tid] = v0 * v0 + v1 * v1;
  __syncthreads();
  for (int st = 128; st > 0; st >>= 1) {
    if (tid < st) red[tid] += red[tid + st];
    __syncthreads();
  }
  if (tid == 0) c2[k] = red[0];
}

// ---------------------------------------------------------------- mean pool over S
__global__ __launch_bounds__(256) void pool_kernel(const float* __restrict__ x,
                                                   float* __restrict__ pooled) {
  int blk = blockIdx.x;
  int b = blk >> 4, ch = blk & 15;            // 16 chunks of 256 rows per batch
  int tid = threadIdx.x;
  const float* base = x + ((long long)b * SS + (long long)ch * 256) * DD;
  float s0 = 0.0f, s1 = 0.0f;
  for (int r = 0; r < 256; ++r) {
    s0 += base[(long long)r * DD + tid];
    s1 += base[(long long)r * DD + tid + 256];
  }
  atomicAdd(&pooled[b * DD + tid], s0);
  atomicAdd(&pooled[b * DD + tid + 256], s1);
}

// ---------------------------------------------------------------- selector MLP + softmax
__global__ __launch_bounds__(256) void mlp_kernel(const float* __restrict__ pooled,
                                                  const float* __restrict__ w1,
                                                  const float* __restrict__ b1,
                                                  const float* __restrict__ w2,
                                                  const float* __restrict__ b2,
                                                  float* __restrict__ wsW,
                                                  float* __restrict__ outW) {
  __shared__ float sH[BB * 256];
  __shared__ float sLg[32];
  int tid = threadIdx.x;
  const float invS = 1.0f / (float)SS;
  for (int b = 0; b < BB; ++b) {
    float acc = b1[tid];
    for (int d = 0; d < DD; ++d)
      acc += (pooled[b * DD + d] * invS) * w1[d * 256 + tid];
    sH[b * 256 + tid] = fmaxf(acc, 0.0f);
  }
  __syncthreads();
  if (tid < 32) {
    int b = tid >> 1, j = tid & 1;
    float acc = b2[j];
    for (int kk = 0; kk < 256; ++kk) acc += sH[b * 256 + kk] * w2[kk * 2 + j];
    sLg[tid] = acc;
  }
  __syncthreads();
  if (tid < BB) {
    float l0 = sLg[tid * 2], l1 = sLg[tid * 2 + 1];
    float mx = fmaxf(l0, l1);
    float e0 = __expf(l0 - mx), e1 = __expf(l1 - mx);
    float inv = 1.0f / (e0 + e1);
    wsW[tid * 2]     = e0 * inv;  outW[tid * 2]     = e0 * inv;
    wsW[tid * 2 + 1] = e1 * inv;  outW[tid * 2 + 1] = e1 * inv;
  }
}

// ---------------------------------------------------------------- fused WMMA quantizer
__global__ __launch_bounds__(512) void fsq_main(const float* __restrict__ x,
                                                const float* __restrict__ codebook,
                                                const float* __restrict__ means,
                                                const float* __restrict__ stds,
                                                const _Float16* __restrict__ cbh,
                                                const float* __restrict__ c2,
                                                const float* __restrict__ wts,
                                                float* __restrict__ out) {
  __shared__ __align__(16) _Float16 sXh[16 * XH_STRIDE];
  __shared__ float sSc[16 * SC_STRIDE];
  __shared__ unsigned long long sRed[16 * 16];
  __shared__ float sC2[KK];
  __shared__ int   sIdx[16];
  __shared__ float sW[2];

  const int tid   = threadIdx.x;                 // 0..511, d = tid
  const int b     = blockIdx.x >> 8;             // 256 s-tiles per batch
  const int stile = blockIdx.x & 255;
  const long long rowBase = (long long)b * SS + (long long)stile * 16;
  const float* xTile = x + rowBase * DD;

  if (tid < KK) sC2[tid] = c2[tid];
  if (tid == 0) { sW[0] = wts[b * 2]; sW[1] = wts[b * 2 + 1]; }

  // stage x tile, normalized, as f16 in LDS
  const float mu  = means[tid];
  const float sg  = stds[tid];
  {
    const float inv = 1.0f / sg;
    #pragma unroll
    for (int r = 0; r < 16; ++r) {
      float xv = xTile[(long long)r * DD + tid];
      sXh[r * XH_STRIDE + tid] = (_Float16)((xv - mu) * inv);
    }
  }
  __syncthreads();

  // ---- distance GEMM: 16 waves, each owns one 16x16 code tile; K = 512 = 16 x 32
  const int lane    = tid & 31;
  const int wv      = tid >> 5;                  // 0..15
  const int nloc    = lane & 15;
  const int hi      = lane >> 4;                 // 0/1 selects K sub-group
  const int codeRow = wv * 16 + nloc;

  v8f acc = {};
  const _Float16* aBase = &sXh[nloc * XH_STRIDE + hi * 8];
  const _Float16* bBase = &cbh[(long long)codeRow * DD + hi * 8];
  #pragma unroll
  for (int ks = 0; ks < 16; ++ks) {
    v8h a0 = *(const v8h*)(aBase + ks * 32);
    v8h a1 = *(const v8h*)(aBase + ks * 32 + 16);
    v8h b0 = *(const v8h*)(bBase + ks * 32);
    v8h b1 = *(const v8h*)(bBase + ks * 32 + 16);
    v16h A = __builtin_shufflevector(a0, a1, 0,1,2,3,4,5,6,7,8,9,10,11,12,13,14,15);
    v16h Bm = __builtin_shufflevector(b0, b1, 0,1,2,3,4,5,6,7,8,9,10,11,12,13,14,15);
    acc = __builtin_amdgcn_wmma_f32_16x16x32_f16(false, A, false, Bm, (short)0, acc,
                                                 false, false);
  }

  // scores: d2-const = c2[k] - 2*x.c   (row norm is argmin-invariant)
  {
    const float cc = sC2[codeRow];
    #pragma unroll
    for (int v = 0; v < 8; ++v) {
      int m = v + 8 * hi;
      sSc[m * SC_STRIDE + codeRow] = cc - 2.0f * acc[v];
    }
  }
  __syncthreads();

  // ---- argmin over 256 codes per row (packed u64 min)
  if (tid < 256) {
    int r = tid & 15, ch = tid >> 4;
    unsigned long long best = ~0ULL;
    #pragma unroll
    for (int i = 0; i < 16; ++i) {
      int code = ch * 16 + i;
      unsigned int ub = __float_as_uint(sSc[r * SC_STRIDE + code]);
      ub = (ub & 0x80000000u) ? ~ub : (ub | 0x80000000u);
      unsigned long long p = ((unsigned long long)ub << 32) | (unsigned int)code;
      if (p < best) best = p;
    }
    sRed[tid] = best;                            // tid = ch*16 + r
  }
  __syncthreads();
  if (tid < 16) {
    unsigned long long best = ~0ULL;
    #pragma unroll
    for (int ch = 0; ch < 16; ++ch) {
      unsigned long long p = sRed[ch * 16 + tid];
      if (p < best) best = p;
    }
    int code = (int)(best & 0xFFFFFFFFu);
    sIdx[tid] = code;
    out[OUT_IDX + rowBase + tid] = (float)code;  // idx_k
  }
  __syncthreads();

  // ---- epilogue: FSQ quantize, gather, weighted combine, commit loss
  const float uw = sW[0], kw = sW[1];
  const float step = 2.0f / 7.0f, invStep = 3.5f;
  float lossAcc = 0.0f;
  #pragma unroll
  for (int r = 0; r < 16; ++r) {
    long long gi = (rowBase + r) * DD + tid;
    float xv = x[gi];
    float xc = fminf(fmaxf(xv, -1.0f), 1.0f);
    float qu = -1.0f + rintf((xc + 1.0f) * invStep) * step;   // round-half-even == jnp.round
    float qk = codebook[sIdx[r] * DD + tid] * sg + mu;
    float du = xv - qu, dk = xv - qk;
    lossAcc += du * du + dk * dk;
    out[OUT_Q + gi] = uw * qu + kw * qk;
  }
  sSc[tid] = lossAcc;                            // reuse score LDS (>= 512 floats)
  __syncthreads();
  for (int st = 256; st > 0; st >>= 1) {
    if (tid < st) sSc[tid] += sSc[tid + st];
    __syncthreads();
  }
  if (tid == 0)
    atomicAdd(&out[OUT_LOSS], sSc[0] * (0.25f / (float)((long long)BB * SS * DD)));
}

// ---------------------------------------------------------------- launcher
extern "C" void kernel_launch(void* const* d_in, const int* in_sizes, int n_in,
                              void* d_out, int out_size, void* d_ws, size_t ws_size,
                              hipStream_t stream) {
  const float* x     = (const float*)d_in[0];
  const float* cb    = (const float*)d_in[1];
  const float* means = (const float*)d_in[2];
  const float* stds  = (const float*)d_in[3];
  const float* w1    = (const float*)d_in[4];
  const float* b1    = (const float*)d_in[5];
  const float* w2    = (const float*)d_in[6];
  const float* b2    = (const float*)d_in[7];
  float* out = (float*)d_out;
  float* wsf = (float*)d_ws;

  float*     pooled = wsf + WS_POOLED;
  float*     c2     = wsf + WS_C2;
  float*     wsW    = wsf + WS_W;
  _Float16*  cbh    = (_Float16*)(wsf + WS_CBH);

  zero_kernel<<<1, 256, 0, stream>>>(pooled, out + OUT_LOSS);
  prep_cb<<<KK, 256, 0, stream>>>(cb, cbh, c2);
  pool_kernel<<<BB * 16, 256, 0, stream>>>(x, pooled);
  mlp_kernel<<<1, 256, 0, stream>>>(pooled, w1, b1, w2, b2, wsW, out + OUT_W);
  fsq_main<<<BB * (SS / 16), 512, 0, stream>>>(x, cb, means, stds, cbh, c2, wsW, out);
}
